// GNNEncoder_48833778156187
// MI455X (gfx1250) — matrix operand order repaired and checked by reading, not compiled, not run on hardware
//
#include <hip/hip_runtime.h>
#include <hip/hip_bf16.h>

// ---------------------------------------------------------------------------
// GAT encoder for gfx1250 (MI455X). wave32. WMMA bf16 16x16x32 for GEMMs.
// ---------------------------------------------------------------------------

typedef __attribute__((ext_vector_type(16))) __bf16 v16bf;
typedef __attribute__((ext_vector_type(8)))  float  v8f;

#define NN     50000
#define EE     640000
#define ETOT   (EE + NN)
#define NTYPES 8
#define SLOPE  0.2f

__device__ __forceinline__ unsigned short f32_to_bf16(float f) {
    unsigned u = __float_as_uint(f);
    unsigned r = u + 0x7FFFu + ((u >> 16) & 1u);
    return (unsigned short)(r >> 16);
}
// order-preserving float<->uint encoding for atomicMax on floats
__device__ __forceinline__ unsigned fenc(float f) {
    unsigned u = __float_as_uint(f);
    return (u & 0x80000000u) ? ~u : (u | 0x80000000u);
}
__device__ __forceinline__ float fdec(unsigned u) {
    return (u & 0x80000000u) ? __uint_as_float(u & 0x7FFFFFFFu)
                             : __uint_as_float(~u);
}

// ---------------------------------------------------------------------------
// fp32 -> bf16 convert
__global__ void k_cvt_bf16(const float* __restrict__ in,
                           unsigned short* __restrict__ out, int n) {
    int i = blockIdx.x * blockDim.x + threadIdx.x;
    if (i < n) out[i] = f32_to_bf16(in[i]);
}

// W[K,128] fp32 -> Wt[128,K] bf16 (transposed so B-operand K runs are contiguous)
__global__ void k_cvt_wt(const float* __restrict__ W,
                         unsigned short* __restrict__ Wt, int K) {
    int i = blockIdx.x * blockDim.x + threadIdx.x;
    if (i >= 128 * K) return;
    int j = i / K, k = i % K;
    Wt[i] = f32_to_bf16(W[k * 128 + j]);
}

// ---------------------------------------------------------------------------
// h[M,128] = A[M,K](bf16) @ B[K,128](bf16, given transposed as Bt[128,K])
// One wave per 16x16 output tile, K stepped by 32 via v_wmma_f32_16x16x32_bf16.
__global__ void k_gemm_wmma(const unsigned short* __restrict__ Abf,
                            const unsigned short* __restrict__ Bt,
                            float* __restrict__ C, int M, int K) {
    int lane = threadIdx.x;                       // 0..31
    int tile = blockIdx.x * blockDim.y + threadIdx.y;
    int tm = tile >> 3;                           // 8 col tiles (128/16)
    int tn = tile & 7;
    if (tm * 16 >= M) return;

    int rowA = tm * 16 + (lane & 15);
    int colB = tn * 16 + (lane & 15);
    int hiA  = (lane >> 4) * 8;                   // K sub-offset per lane half
    int hiB  = (lane >> 4) * 8;

    v8f acc = {};
    union U { uint4 q[2]; v16bf v; };
    U a, b;
    for (int kk = 0; kk < K; kk += 32) {
        const unsigned short* ap = Abf + (size_t)rowA * K + kk + hiA;
        a.q[0] = *(const uint4*)(ap);             // elems 0..7  -> K = kk+hiA..+7
        a.q[1] = *(const uint4*)(ap + 16);        // elems 8..15 -> K = kk+16+hiA..
        const unsigned short* bp = Bt + (size_t)colB * K + kk + hiB;
        b.q[0] = *(const uint4*)(bp);             // elems 0..7  -> K = kk+hiB..+7
        b.q[1] = *(const uint4*)(bp + 16);
        acc = __builtin_amdgcn_wmma_f32_16x16x32_bf16(
                  false, a.v, false, b.v, (short)0, acc, false, false);
    }
    int rbase = tm * 16 + ((lane >> 4) ? 8 : 0);
    int c0    = tn * 16 + (lane & 15);
#pragma unroll
    for (int r = 0; r < 8; ++r)
        C[(size_t)(rbase + r) * 128 + c0] = acc[r];
}

// ---------------------------------------------------------------------------
// incoming-edge type histogram (for self-loop fill_value='mean')
__global__ void k_hist(const int* __restrict__ dst, const int* __restrict__ attr,
                       float* __restrict__ hist) {
    int i = blockIdx.x * blockDim.x + threadIdx.x;
    if (i < EE) atomicAdd(&hist[dst[i] * NTYPES + attr[i]], 1.0f);
}

// Wea[k,h] = sum_c We[k,h*C+c]*att_e[h,c];  aetype[t,h] = edge_table[t,:] @ Wea[:,h]
__global__ void k_prep_edge(const float* __restrict__ We,
                            const float* __restrict__ att_e,
                            const float* __restrict__ etab,
                            float* __restrict__ Wea, float* __restrict__ aetype,
                            int H, int C) {
    int t = threadIdx.x;
    for (int i = t; i < 32 * H; i += blockDim.x) {
        int k = i / H, h = i % H;
        float s = 0.f;
        for (int c = 0; c < C; ++c) s += We[k * 128 + h * C + c] * att_e[h * C + c];
        Wea[i] = s;
    }
    __threadfence_block();
    __syncthreads();
    for (int i = t; i < NTYPES * H; i += blockDim.x) {
        int ty = i / H, h = i % H;
        float s = 0.f;
        for (int k = 0; k < 32; ++k) s += etab[ty * 32 + k] * Wea[k * H + h];
        aetype[i] = s;
    }
}

// per-node attention scalars a_src/a_dst
__global__ void k_att_nodes(const float* __restrict__ h,
                            const float* __restrict__ att_src,
                            const float* __restrict__ att_dst,
                            float* __restrict__ a_src, float* __restrict__ a_dst,
                            int H, int C) {
    int i = blockIdx.x * blockDim.x + threadIdx.x;
    if (i >= NN * H) return;
    int n = i / H, hh = i % H;
    const float* hp = h + (size_t)n * 128 + hh * C;
    const float* as = att_src + hh * C;
    const float* ad = att_dst + hh * C;
    float s = 0.f, d = 0.f;
    for (int c = 0; c < C; ++c) { float v = hp[c]; s += v * as[c]; d += v * ad[c]; }
    a_src[i] = s; a_dst[i] = d;
}

// self-loop attention term: (hist @ aetype) / max(deg,1)
__global__ void k_aeloop(const float* __restrict__ hist,
                         const float* __restrict__ aetype,
                         float* __restrict__ aeloop, int H) {
    int i = blockIdx.x * blockDim.x + threadIdx.x;
    if (i >= NN * H) return;
    int n = i / H, h = i % H;
    float deg = 0.f, s = 0.f;
    for (int t = 0; t < NTYPES; ++t) {
        float c = hist[n * NTYPES + t];
        deg += c; s += c * aetype[t * H + h];
    }
    aeloop[i] = s / fmaxf(deg, 1.0f);
}

// alpha = leakyrelu(a_src[s]+a_dst[d]+a_e); segment max via encoded atomicMax
__global__ void k_alpha(const int* __restrict__ src, const int* __restrict__ dst,
                        const int* __restrict__ attr,
                        const float* __restrict__ a_src, const float* __restrict__ a_dst,
                        const float* __restrict__ aetype, const float* __restrict__ aeloop,
                        float* __restrict__ alpha, unsigned* __restrict__ amax, int H) {
    int id = blockIdx.x * blockDim.x + threadIdx.x;
    if (id >= ETOT) return;
    int s, d; const float* ae;
    if (id < EE) { s = src[id]; d = dst[id]; ae = aetype + attr[id] * H; }
    else         { s = d = id - EE;          ae = aeloop + (size_t)(id - EE) * H; }
    for (int h = 0; h < H; ++h) {
        float a = a_src[s * H + h] + a_dst[d * H + h] + ae[h];
        a = (a > 0.f) ? a : SLOPE * a;
        alpha[(size_t)id * H + h] = a;
        atomicMax(&amax[d * H + h], fenc(a));
    }
}

// ex = exp(alpha - amax[d]); denom[d] += ex
__global__ void k_expden(const int* __restrict__ dst, float* __restrict__ alpha,
                         const unsigned* __restrict__ amax,
                         float* __restrict__ denom, int H) {
    int id = blockIdx.x * blockDim.x + threadIdx.x;
    if (id >= ETOT) return;
    int d = (id < EE) ? dst[id] : id - EE;
    for (int h = 0; h < H; ++h) {
        float m  = fdec(amax[d * H + h]);
        float ex = __expf(alpha[(size_t)id * H + h] - m);
        alpha[(size_t)id * H + h] = ex;
        atomicAdd(&denom[d * H + h], ex);
    }
}

// out[dst] += h[src] * softmax-weight ; one wave32 per edge, 4 channels/lane
__global__ void k_agg(const int* __restrict__ src, const int* __restrict__ dst,
                      const float* __restrict__ hfeat, const float* __restrict__ alpha,
                      const float* __restrict__ denom, float* __restrict__ out,
                      int H, int C) {
    int lane = threadIdx.x & 31;
    int id = blockIdx.x * (blockDim.x >> 5) + (threadIdx.x >> 5);
    if (id >= ETOT) return;
    int s, d;
    if (id < EE) { s = src[id]; d = dst[id]; } else { s = d = id - EE; }
#pragma unroll
    for (int i = 0; i < 4; ++i) {
        int j = i * 32 + lane;
        int h = j / C;
        float w = alpha[(size_t)id * H + h] / (denom[d * H + h] + 1e-16f);
        atomicAdd(&out[(size_t)d * 128 + j], hfeat[(size_t)s * 128 + j] * w);
    }
}

// epilogue: +bias, optional relu, optional residual; fp32 out + bf16 copy
__global__ void k_epi(const float* __restrict__ acc, const float* __restrict__ bias,
                      const float* __restrict__ resid, float* __restrict__ outf,
                      unsigned short* __restrict__ outbf, int doRelu) {
    int i = blockIdx.x * blockDim.x + threadIdx.x;
    if (i >= NN * 128) return;
    float v = acc[i] + bias[i & 127];
    if (doRelu) v = fmaxf(v, 0.f);
    if (resid) v += resid[i];
    outf[i] = v;
    if (outbf) outbf[i] = f32_to_bf16(v);
}

// ---------------------------------------------------------------------------
extern "C" void kernel_launch(void* const* d_in, const int* in_sizes, int n_in,
                              void* d_out, int out_size, void* d_ws, size_t ws_size,
                              hipStream_t stream) {
    (void)in_sizes; (void)n_in; (void)out_size; (void)ws_size;
    const float* x    = (const float*)d_in[0];
    const int*   ei   = (const int*)d_in[1];
    const int*   batch= (const int*)d_in[2];
    const int*   attr = (const int*)d_in[3];
    const float* etab = (const float*)d_in[4];
    const int* src = ei;
    const int* dst = ei + EE;

    // workspace carve-up
    char* w = (char*)d_ws;
    auto alloc = [&](size_t bytes) -> void* {
        void* p = (void*)w; w += (bytes + 255) & ~(size_t)255; return p;
    };
    unsigned short* xbf  = (unsigned short*)alloc((size_t)NN * 128 * 2);
    unsigned short* Wt   = (unsigned short*)alloc(128 * 128 * 2);
    float* hbuf   = (float*)alloc((size_t)NN * 128 * 4);
    float* obuf   = (float*)alloc((size_t)NN * 128 * 4);
    float* bufA   = (float*)alloc((size_t)NN * 128 * 4);
    float* bufB   = (float*)alloc((size_t)NN * 128 * 4);
    float* a_src  = (float*)alloc((size_t)NN * 8 * 4);
    float* a_dst  = (float*)alloc((size_t)NN * 8 * 4);
    unsigned* amax= (unsigned*)alloc((size_t)NN * 8 * 4);
    float* denom  = (float*)alloc((size_t)NN * 8 * 4);
    float* alpha  = (float*)alloc((size_t)ETOT * 8 * 4);
    float* hist   = (float*)alloc((size_t)NN * NTYPES * 4);
    float* aeloop = (float*)alloc((size_t)NN * 8 * 4);
    float* Wea    = (float*)alloc(32 * 8 * 4);
    float* aetype = (float*)alloc(NTYPES * 8 * 4);

    // one-time: incoming edge-type histogram
    hipMemsetAsync(hist, 0, (size_t)NN * NTYPES * 4, stream);
    k_hist<<<(EE + 255) / 256, 256, 0, stream>>>(dst, attr, hist);
    // one-time: input features -> bf16 (row length 64)
    k_cvt_bf16<<<(NN * 64 + 255) / 256, 256, 0, stream>>>(x, xbf, NN * 64);

    auto layer = [&](int K, int H, int C, const float* const* P,
                     const float* resid, float* xout_f,
                     unsigned short* xout_bf, int doRelu) {
        const float* Wp   = P[0];
        const float* atts = P[1];
        const float* attd = P[2];
        const float* Wep  = P[3];
        const float* atte = P[4];
        const float* bias = P[5];
        // weights -> transposed bf16
        k_cvt_wt<<<(128 * K + 255) / 256, 256, 0, stream>>>(Wp, Wt, K);
        // h = x @ W  (WMMA)
        dim3 gblk(32, 4);
        int tiles = (NN / 16) * 8;
        k_gemm_wmma<<<(tiles + 3) / 4, gblk, 0, stream>>>(xbf, Wt, hbuf, NN, K);
        // attention tables + per-node scalars
        k_prep_edge<<<1, 256, 0, stream>>>(Wep, atte, etab, Wea, aetype, H, C);
        k_att_nodes<<<(NN * H + 255) / 256, 256, 0, stream>>>(hbuf, atts, attd,
                                                              a_src, a_dst, H, C);
        k_aeloop<<<(NN * H + 255) / 256, 256, 0, stream>>>(hist, aetype, aeloop, H);
        // segment softmax + aggregate
        hipMemsetAsync(amax,  0, (size_t)NN * H * 4, stream);
        hipMemsetAsync(denom, 0, (size_t)NN * H * 4, stream);
        hipMemsetAsync(obuf,  0, (size_t)NN * 128 * 4, stream);
        k_alpha<<<(ETOT + 255) / 256, 256, 0, stream>>>(src, dst, attr, a_src, a_dst,
                                                        aetype, aeloop, alpha, amax, H);
        k_expden<<<(ETOT + 255) / 256, 256, 0, stream>>>(dst, alpha, amax, denom, H);
        k_agg<<<(ETOT + 7) / 8, 256, 0, stream>>>(src, dst, hbuf, alpha, denom,
                                                  obuf, H, C);
        // epilogue
        k_epi<<<(NN * 128 + 255) / 256, 256, 0, stream>>>(obuf, bias, resid,
                                                          xout_f, xout_bf, doRelu);
    };

    const float* P1[6] = {(const float*)d_in[5],  (const float*)d_in[6],
                          (const float*)d_in[7],  (const float*)d_in[8],
                          (const float*)d_in[9],  (const float*)d_in[10]};
    const float* P2[6] = {(const float*)d_in[11], (const float*)d_in[12],
                          (const float*)d_in[13], (const float*)d_in[14],
                          (const float*)d_in[15], (const float*)d_in[16]};
    const float* P3[6] = {(const float*)d_in[17], (const float*)d_in[18],
                          (const float*)d_in[19], (const float*)d_in[20],
                          (const float*)d_in[21], (const float*)d_in[22]};
    const float* P4[6] = {(const float*)d_in[23], (const float*)d_in[24],
                          (const float*)d_in[25], (const float*)d_in[26],
                          (const float*)d_in[27], (const float*)d_in[28]};

    float* xout = (float*)d_out;
    layer( 64, 8,  16, P1, nullptr, bufA, xbf, 1);   // x1 = relu(gat(x))
    layer(128, 8,  16, P2, bufA,    bufB, xbf, 1);   // x2 = relu(gat(x1)) + x1
    layer(128, 8,  16, P3, bufB,    bufA, xbf, 1);   // x3 = relu(gat(x2)) + x2
    layer(128, 1, 128, P4, bufA,    xout, nullptr, 0); // x4 = gat(x3) + x3

    // batch passthrough (int bits) after the N*128 float features
    hipMemcpyAsync((char*)d_out + (size_t)NN * 128 * 4, (const void*)batch,
                   (size_t)NN * 4, hipMemcpyDeviceToDevice, stream);
}
